// MMPNType_57647051047687
// MI455X (gfx1250) — compile-verified, matrix operand order checked by reading, not run on hardware
//
#include <hip/hip_runtime.h>
#include <hip/hip_bf16.h>

// ---------------- problem constants (match reference) ----------------
#define NB    512          // graphs
#define NPG   64           // nodes per graph
#define EPG   256          // edges per graph
#define NN    (NB*NPG)     // 32768 nodes
#define NE    (NB*EPG)     // 131072 edges
#define FN    128
#define FE    64
#define FG    64
#define MH    512
#define MD    256
#define UH    512
#define UD    256
#define GD    128
#define AD    8

#define K_MSG (2*FN+FE+FG) // 384
#define K_UPD (MD+FN+FG)   // 448

#define ROWS  64           // rows (edges/nodes) per workgroup = 4 M-tiles

// ---------------- WMMA types ----------------
typedef __attribute__((ext_vector_type(16))) __bf16        v16bf;
typedef __attribute__((ext_vector_type(8)))  float         v8f;
typedef __attribute__((ext_vector_type(4)))  unsigned int  v4u;

union FragU {
    v16bf v;
    unsigned short u16[16];
    unsigned int   u32[8];
    v4u            u128[2];
};

__device__ __forceinline__ unsigned short f32_to_bf16(float f) {
    unsigned int u = __float_as_uint(f);
    u += 0x7FFFu + ((u >> 16) & 1u);     // round-to-nearest-even
    return (unsigned short)(u >> 16);
}

__device__ __forceinline__ v8f wmma_bf16(const FragU& a, const FragU& b, v8f c) {
    return __builtin_amdgcn_wmma_f32_16x16x32_bf16(
               false, a.v, false, b.v, (short)0, c, false, false);
}

// ---------------- weight pre-pack: fp32 [K][N] -> fragment-major bf16 ----------------
// Layout: [col_tile][k_block][lane(32)][16 bf16]; per ISA 7.12.2 (16-bit B 32x16):
//   n = ct*16 + (lane&15), h = lane>>4, u16[2v+j] = B[k0 + (v>>2)*16 + h*8 + (v&3)*2 + j][n]
__global__ __launch_bounds__(256) void pack_b_kernel(const float* __restrict__ W,
                                                     unsigned short* __restrict__ out,
                                                     int K, int Nfull) {
    int idx = blockIdx.x * 256 + threadIdx.x;
    int total = K * Nfull;
    if (idx >= total) return;
    int i    = idx & 15;
    int lane = (idx >> 4) & 31;
    int blk  = idx >> 9;                 // = ct*KB + k0b
    int KB   = K >> 5;
    int ct   = blk / KB;
    int k0b  = blk - ct * KB;
    int v = i >> 1, j = i & 1, h = lane >> 4;
    int k = (k0b << 5) + ((v >> 2) << 4) + (h << 3) + ((v & 3) << 1) + j;
    int n = (ct << 4) + (lane & 15);
    out[idx] = f32_to_bf16(W[(size_t)k * Nfull + n]);
}

// ---------------- register-blocked GEMM: 4 M-tiles x 2 N-tiles per wave ----------------
// A: bf16 LDS tile [64][ldx] row-major (16B-aligned rows). B: packed bf16.
// Each B fragment from L2 feeds 4 WMMAs; each A fragment (2x ds_load_b128) feeds 2.
template <int KB>
__device__ __forceinline__ void gemm_block_4x2(
        const unsigned short* __restrict__ Xs, int ldx,
        const unsigned short* __restrict__ Wp, int ct0, int ct1,
        int lane, v8f acc[4][2]) {
    const int m = lane & 15, h = lane >> 4;
    const unsigned short* xrow = Xs + m * ldx + (h << 3);
    const unsigned short* wp0 = Wp + (((size_t)ct0 * KB) * 32 + lane) * 16;
    const unsigned short* wp1 = Wp + (((size_t)ct1 * KB) * 32 + lane) * 16;
#pragma unroll
    for (int kb = 0; kb < KB; ++kb) {
        FragU fb0, fb1;
        fb0.v = *(const v16bf*)(wp0 + (size_t)kb * 512);   // 32B contiguous
        fb1.v = *(const v16bf*)(wp1 + (size_t)kb * 512);
        FragU fa[4];
#pragma unroll
        for (int mt = 0; mt < 4; ++mt) {
            const unsigned short* xp = xrow + (size_t)(mt * 16) * ldx + (kb << 5);
            fa[mt].u128[0] = *(const v4u*)(xp);            // ds_load_b128
            fa[mt].u128[1] = *(const v4u*)(xp + 16);       // ds_load_b128
        }
#pragma unroll
        for (int mt = 0; mt < 4; ++mt) {
            acc[mt][0] = wmma_bf16(fa[mt], fb0, acc[mt][0]);
            acc[mt][1] = wmma_bf16(fa[mt], fb1, acc[mt][1]);
        }
    }
}

// ---------------- zero-init aggr ----------------
__global__ __launch_bounds__(256) void zero_f4_kernel(float4* __restrict__ p, int n4) {
    int i = blockIdx.x * 256 + threadIdx.x;
    if (i < n4) p[i] = make_float4(0.f, 0.f, 0.f, 0.f);
}

// ---------------- fused message MLP + scatter-max ----------------
// one workgroup (256 thr = 8 wave32) per 64-edge tile
#define LDX_MSG (K_MSG + 8)   // 392 -> 784B rows (16B aligned, bank-conflict free)
#define LDH     (MH + 8)      // 520 -> 1040B rows

__global__ __launch_bounds__(256) void msg_kernel(
        const float* __restrict__ nodes, const float* __restrict__ edge_attr,
        const float* __restrict__ gattr,
        const int* __restrict__ src, const int* __restrict__ tgt,
        const int* __restrict__ egid,
        const unsigned short* __restrict__ Wm0p, const float* __restrict__ bm0,
        const unsigned short* __restrict__ Wm1p, const float* __restrict__ bm1,
        float* __restrict__ aggr) {
    __shared__ unsigned short Xs[ROWS * LDX_MSG];
    __shared__ unsigned short Hs[ROWS * LDH];
    __shared__ int tgts[ROWS];

    const int e0 = blockIdx.x * ROWS;
    const int t = threadIdx.x;

    // ---- stage concat input tile as bf16: [src(128) | edge(64) | tgt(128) | glob(64)]
    {
        const int r0 = t >> 4, l16 = t & 15;
        if (t < ROWS) tgts[t] = tgt[e0 + t];
#pragma unroll
        for (int rr = 0; rr < ROWS / 16; ++rr) {
            const int row = r0 + rr * 16;
            const int e = e0 + row;
            const int s = src[e], d = tgt[e], g = egid[e];
            for (int c = l16; c < K_MSG; c += 16) {
                float f;
                if      (c < FN)          f = nodes[(size_t)s * FN + c];
                else if (c < FN + FE)     f = edge_attr[(size_t)e * FE + (c - FN)];
                else if (c < 2*FN + FE)   f = nodes[(size_t)d * FN + (c - FN - FE)];
                else                      f = gattr[(size_t)g * FG + (c - 2*FN - FE)];
                Xs[row * LDX_MSG + c] = f32_to_bf16(f);
            }
        }
    }
    __syncthreads();

    const int wave = t >> 5, lane = t & 31;
    const int m = lane & 15, h = lane >> 4;

    // ---- layer 0: 384 -> 512, ReLU -> LDS   (32 col tiles: 2 passes of [wave, wave+8] pairs)
#pragma unroll
    for (int p = 0; p < 2; ++p) {
        const int ct0 = wave + p * 16, ct1 = ct0 + 8;
        v8f acc[4][2] = {};
        gemm_block_4x2<K_MSG / 32>(Xs, LDX_MSG, Wm0p, ct0, ct1, lane, acc);
#pragma unroll
        for (int mt = 0; mt < 4; ++mt) {
#pragma unroll
            for (int c = 0; c < 2; ++c) {
                const int ct = c ? ct1 : ct0;
                const float bias = bm0[ct * 16 + m];
#pragma unroll
                for (int r = 0; r < 8; ++r) {
                    float vv = fmaxf(acc[mt][c][r] + bias, 0.f);
                    Hs[(mt * 16 + r + h * 8) * LDH + ct * 16 + m] = f32_to_bf16(vv);
                }
            }
        }
    }
    __syncthreads();

    // ---- layer 1: 512 -> 256, ReLU, scatter-max into aggr (16 col tiles: 1 pass)
    {
        const int ct0 = wave, ct1 = wave + 8;
        v8f acc[4][2] = {};
        gemm_block_4x2<MH / 32>(Hs, LDH, Wm1p, ct0, ct1, lane, acc);
#pragma unroll
        for (int mt = 0; mt < 4; ++mt) {
#pragma unroll
            for (int c = 0; c < 2; ++c) {
                const int ct = c ? ct1 : ct0;
                const float bias = bm1[ct * 16 + m];
#pragma unroll
                for (int r = 0; r < 8; ++r) {
                    float vv = fmaxf(acc[mt][c][r] + bias, 0.f);
                    const int node = tgts[mt * 16 + r + h * 8];
                    // vv >= 0 and aggr initialized to 0 => int-bit max == float max
                    atomicMax((int*)&aggr[(size_t)node * MD + ct * 16 + m],
                              __float_as_int(vv));
                }
            }
        }
    }
}

// ---------------- fused node-update MLP ----------------
#define LDX_UPD (K_UPD + 8)   // 456 -> 912B rows

__global__ __launch_bounds__(256) void upd_kernel(
        const float* __restrict__ nodes, const float* __restrict__ aggr,
        const float* __restrict__ gattr, const int* __restrict__ bidx,
        const unsigned short* __restrict__ Wu0p, const float* __restrict__ bu0,
        const unsigned short* __restrict__ Wu1p, const float* __restrict__ bu1,
        float* __restrict__ upd) {
    __shared__ unsigned short Xs[ROWS * LDX_UPD];
    __shared__ unsigned short Hs[ROWS * LDH];

    const int n0row = blockIdx.x * ROWS;
    const int t = threadIdx.x;

    // ---- stage concat input tile: [nodes(128) | aggr(256) | glob(64)]
    {
        const int r0 = t >> 4, l16 = t & 15;
#pragma unroll
        for (int rr = 0; rr < ROWS / 16; ++rr) {
            const int row = r0 + rr * 16;
            const int nd = n0row + row;
            const int g = bidx[nd];
            for (int c = l16; c < K_UPD; c += 16) {
                float f;
                if      (c < FN)       f = nodes[(size_t)nd * FN + c];
                else if (c < FN + MD)  f = aggr[(size_t)nd * MD + (c - FN)];
                else                   f = gattr[(size_t)g * FG + (c - FN - MD)];
                Xs[row * LDX_UPD + c] = f32_to_bf16(f);
            }
        }
    }
    __syncthreads();

    const int wave = t >> 5, lane = t & 31;
    const int m = lane & 15, h = lane >> 4;

    // ---- layer 0: 448 -> 512, ReLU -> LDS
#pragma unroll
    for (int p = 0; p < 2; ++p) {
        const int ct0 = wave + p * 16, ct1 = ct0 + 8;
        v8f acc[4][2] = {};
        gemm_block_4x2<K_UPD / 32>(Xs, LDX_UPD, Wu0p, ct0, ct1, lane, acc);
#pragma unroll
        for (int mt = 0; mt < 4; ++mt) {
#pragma unroll
            for (int c = 0; c < 2; ++c) {
                const int ct = c ? ct1 : ct0;
                const float bias = bu0[ct * 16 + m];
#pragma unroll
                for (int r = 0; r < 8; ++r) {
                    float vv = fmaxf(acc[mt][c][r] + bias, 0.f);
                    Hs[(mt * 16 + r + h * 8) * LDH + ct * 16 + m] = f32_to_bf16(vv);
                }
            }
        }
    }
    __syncthreads();

    // ---- layer 1: 512 -> 256, ReLU -> upd (global)
    {
        const int ct0 = wave, ct1 = wave + 8;
        v8f acc[4][2] = {};
        gemm_block_4x2<UH / 32>(Hs, LDH, Wu1p, ct0, ct1, lane, acc);
#pragma unroll
        for (int mt = 0; mt < 4; ++mt) {
#pragma unroll
            for (int c = 0; c < 2; ++c) {
                const int ct = c ? ct1 : ct0;
                const float bias = bu1[ct * 16 + m];
#pragma unroll
                for (int r = 0; r < 8; ++r) {
                    float vv = fmaxf(acc[mt][c][r] + bias, 0.f);
                    upd[(size_t)(n0row + mt * 16 + r + h * 8) * UD + ct * 16 + m] = vv;
                }
            }
        }
    }
}

// ---------------- graph pooling + group MLP + action head + softmax ----------------
__global__ __launch_bounds__(128) void head_kernel(
        const float* __restrict__ upd, const float* __restrict__ gattr,
        const int* __restrict__ tbw,
        const float* __restrict__ Wg, const float* __restrict__ bg,
        const float* __restrict__ Wa, const float* __restrict__ ba,
        float* __restrict__ out) {
    __shared__ float agg[UD];
    __shared__ float grp[GD];
    __shared__ float logits[AD];

    const int b = blockIdx.x;
    const int t = threadIdx.x;

    // segment-max over the graph's 64 nodes (upd >= 0, clamp-at-0 matches reference)
    for (int c = t; c < UD; c += 128) {
        float mx = 0.f;
        const float* base = upd + (size_t)b * NPG * UD + c;
        for (int j = 0; j < NPG; ++j) mx = fmaxf(mx, base[(size_t)j * UD]);
        agg[c] = mx;
    }
    __syncthreads();

    // group = relu([agg | global] @ Wg + bg)   (320 -> 128)
    {
        float s = bg[t];
        for (int k = 0; k < UD; ++k) s += agg[k] * Wg[(size_t)k * GD + t];
        for (int k = 0; k < FG; ++k) s += gattr[(size_t)b * FG + k] * Wg[(size_t)(UD + k) * GD + t];
        grp[t] = fmaxf(s, 0.f);
    }
    __syncthreads();

    // logits = [upd[chosen] | group] @ Wa + ba   (384 -> 8)
    if (t < AD) {
        const float* ch = upd + (size_t)tbw[b] * UD;
        float s = ba[t];
        for (int k = 0; k < UD; ++k) s += ch[k] * Wa[(size_t)k * AD + t];
        for (int k = 0; k < GD; ++k) s += grp[k] * Wa[(size_t)(UD + k) * AD + t];
        logits[t] = s;
    }
    __syncthreads();

    if (t == 0) {
        float mx = logits[0];
        for (int i = 1; i < AD; ++i) mx = fmaxf(mx, logits[i]);
        float ex[AD], sum = 0.f;
        for (int i = 0; i < AD; ++i) { ex[i] = __expf(logits[i] - mx); sum += ex[i]; }
        float inv = 1.f / sum;
        for (int i = 0; i < AD; ++i) out[(size_t)b * AD + i] = ex[i] * inv;
    }
}

// ---------------- launcher ----------------
extern "C" void kernel_launch(void* const* d_in, const int* in_sizes, int n_in,
                              void* d_out, int out_size, void* d_ws, size_t ws_size,
                              hipStream_t stream) {
    const float* nodes     = (const float*)d_in[0];
    const float* edge_attr = (const float*)d_in[1];
    const float* gattr     = (const float*)d_in[2];
    const int*   src       = (const int*)d_in[3];
    const int*   tgt       = (const int*)d_in[4];
    const int*   egid      = (const int*)d_in[5];
    const int*   bidx      = (const int*)d_in[6];
    const int*   tbw       = (const int*)d_in[7];
    const float* Wm0 = (const float*)d_in[8];   const float* bm0 = (const float*)d_in[9];
    const float* Wm1 = (const float*)d_in[10];  const float* bm1 = (const float*)d_in[11];
    const float* Wu0 = (const float*)d_in[12];  const float* bu0 = (const float*)d_in[13];
    const float* Wu1 = (const float*)d_in[14];  const float* bu1 = (const float*)d_in[15];
    const float* Wg  = (const float*)d_in[16];  const float* bg  = (const float*)d_in[17];
    const float* Wa  = (const float*)d_in[18];  const float* ba  = (const float*)d_in[19];

    char* ws = (char*)d_ws;
    float* aggr = (float*)ws;            ws += (size_t)NN * MD * sizeof(float);   // 33.5 MB
    float* upd  = (float*)ws;            ws += (size_t)NN * UD * sizeof(float);   // 33.5 MB
    unsigned short* Wm0p = (unsigned short*)ws;  ws += (size_t)K_MSG * MH * 2;
    unsigned short* Wm1p = (unsigned short*)ws;  ws += (size_t)MH * MD * 2;
    unsigned short* Wu0p = (unsigned short*)ws;  ws += (size_t)K_UPD * UH * 2;
    unsigned short* Wu1p = (unsigned short*)ws;  ws += (size_t)UH * UD * 2;

    // zero aggr (scatter-max identity = 0 for ReLU outputs)
    {
        int n4 = NN * MD / 4;
        zero_f4_kernel<<<(n4 + 255) / 256, 256, 0, stream>>>((float4*)aggr, n4);
    }
    // pre-pack weights into WMMA B-fragment-major bf16
    pack_b_kernel<<<(K_MSG * MH + 255) / 256, 256, 0, stream>>>(Wm0, Wm0p, K_MSG, MH);
    pack_b_kernel<<<(MH * MD + 255) / 256, 256, 0, stream>>>(Wm1, Wm1p, MH, MD);
    pack_b_kernel<<<(K_UPD * UH + 255) / 256, 256, 0, stream>>>(Wu0, Wu0p, K_UPD, UH);
    pack_b_kernel<<<(UH * UD + 255) / 256, 256, 0, stream>>>(Wu1, Wu1p, UH, UD);

    // fused message MLP + scatter-max
    msg_kernel<<<NE / ROWS, 256, 0, stream>>>(nodes, edge_attr, gattr, src, tgt, egid,
                                              Wm0p, bm0, Wm1p, bm1, aggr);
    // fused node-update MLP
    upd_kernel<<<NN / ROWS, 256, 0, stream>>>(nodes, aggr, gattr, bidx,
                                              Wu0p, bu0, Wu1p, bu1, upd);
    // pooling + group + action + softmax
    head_kernel<<<NB, 128, 0, stream>>>(upd, gattr, tbw, Wg, bg, Wa, ba, (float*)d_out);
}
